// asap_45286135169788
// MI455X (gfx1250) — matrix operand order repaired and checked by reading, not compiled, not run on hardware
//
#include <hip/hip_runtime.h>

// ---------------- fixed problem dimensions (from the reference) ----------------
#define N__  2048
#define E__  32768
#define ET__ (E__ + N__)     // edges + self loops
#define IN__ 128
#define H__  512
#define O__  256
#define K1__ 1024            // N/2
#define K2__ 512             // K1/2

// GEMM macro-tile config: 256 threads (8 waves), 64x64 C tile, K panels of 128
#define TM 64
#define TN 64
#define TK 128
#define TKP (TK + 8)         // +16B row padding to rotate LDS banks

#define USE_ASYNC_LDS 1      // 1: GLOBAL_LOAD_ASYNC_TO_LDS_B128 (ASYNCcnt); 0: VGPR staging

typedef __attribute__((ext_vector_type(16))) __bf16 v16bf;
typedef __attribute__((ext_vector_type(8)))  float  v8f;

// ---------------- small device helpers ----------------
__device__ __forceinline__ __bf16 f2b(float x) {
  unsigned u = __float_as_uint(x);
  unsigned r = (u + 0x7FFFu + ((u >> 16) & 1u)) >> 16;   // round-to-nearest-even
  unsigned short h = (unsigned short)r;
  return *reinterpret_cast<__bf16*>(&h);
}
__device__ __forceinline__ float b2f(__bf16 x) {
  unsigned short h = *reinterpret_cast<unsigned short*>(&x);
  return __uint_as_float(((unsigned)h) << 16);
}
__device__ __forceinline__ unsigned fkey(float x) {   // order-preserving float->uint
  unsigned u = __float_as_uint(x);
  return (u & 0x80000000u) ? ~u : (u | 0x80000000u);
}
__device__ __forceinline__ float funkey(unsigned k) {
  unsigned u = (k & 0x80000000u) ? (k ^ 0x80000000u) : ~k;
  return __uint_as_float(u);
}
__device__ __forceinline__ float lrelu(float x) { return x > 0.f ? x : 0.2f * x; }
__device__ __forceinline__ float sigm(float x)  { return 1.f / (1.f + expf(-x)); }
__device__ __forceinline__ void edge_sd(const int* ei, int e, int& s, int& d) {
  if (e < E__) { s = ei[e]; d = ei[E__ + e]; } else { s = d = e - E__; }
}

// ---- async global->LDS copy (16B per lane). LDS offset = low 32 bits of generic ptr. ----
__device__ __forceinline__ void cp_b128(__bf16* lds, const __bf16* g) {
#if USE_ASYNC_LDS
  unsigned lo = (unsigned)(size_t)lds;
  unsigned long long ga = (unsigned long long)(size_t)g;
  asm volatile("global_load_async_to_lds_b128 %0, %1, off"
               :: "v"(lo), "v"(ga) : "memory");
#else
  *reinterpret_cast<uint4*>(lds) = *reinterpret_cast<const uint4*>(g);
#endif
}
__device__ __forceinline__ void cp_wait() {
#if USE_ASYNC_LDS
  asm volatile("s_wait_asynccnt 0" ::: "memory");
#endif
}

// ---------------- fragment gathers (generic pointers; LDS-resident -> ds_load_b128) ----------------
__device__ __forceinline__ v16bf frag_a(const __bf16* p) {
  // 16-bit A 16x32 layout: this lane holds K {0..7} and {16..23} (hi half shifts by 8)
  v16bf f;
#pragma unroll
  for (int i = 0; i < 8; ++i) f[i] = p[i];
#pragma unroll
  for (int i = 0; i < 8; ++i) f[8 + i] = p[16 + i];
  return f;
}
__device__ __forceinline__ v16bf frag_b(const __bf16* p) {
  // 16-bit B 32x16 layout: same K order as A: K {0..7} and {16..23} (hi half shifts by 8)
  v16bf f;
#pragma unroll
  for (int i = 0; i < 8; ++i) f[i] = p[i];
#pragma unroll
  for (int i = 0; i < 8; ++i) f[8 + i] = p[16 + i];
  return f;
}

// ---------------- WMMA GEMM: C[M,Nc] = A[M,K](bf16,row) x B (given as Bt[Nc,K]) + bias ----------------
// block = 256 threads (8 waves); macro-tile 64x64; wave (rt,ct) owns rows rt*16.., cols ct*32..
__global__ __launch_bounds__(256) void k_wmma_gemm(
    const __bf16* __restrict__ A, const __bf16* __restrict__ Bt,
    float* __restrict__ C, const float* __restrict__ bias,
    int M, int Nc, int K) {
  __shared__ __bf16 As[TM][TKP];
  __shared__ __bf16 Bs[TN][TKP];
  const int tid  = threadIdx.x;
  const int lane = tid & 31;
  const int wave = tid >> 5;
  const int low = lane & 15, hi = lane >> 4;
  const int rt = wave >> 1;                 // 0..3
  const int ct = wave & 1;                  // 0..1
  const int gcol0 = blockIdx.x * TN;
  const int grow0 = blockIdx.y * TM;
  // staging split: 4 lanes per row, each lane covers 32 consecutive k (4 x 16B)
  const int sr = tid >> 2;                  // 0..63
  const int sk = (tid & 3) * 32;            // 0,32,64,96

  v8f acc0 = {}; v8f acc1 = {};
  for (int k0 = 0; k0 < K; k0 += TK) {
    const __bf16* ga = A  + (size_t)(grow0 + sr) * K + k0 + sk;
    const __bf16* gb = Bt + (size_t)(gcol0 + sr) * K + k0 + sk;
#pragma unroll
    for (int q = 0; q < 4; ++q) {
      cp_b128(&As[sr][sk + q * 8], ga + q * 8);
      cp_b128(&Bs[sr][sk + q * 8], gb + q * 8);
    }
    cp_wait();
    __syncthreads();
#pragma unroll
    for (int u = 0; u < 4; ++u) {
      const int kk = u * 32;
      v16bf a  = frag_a(&As[rt * 16 + low][kk + hi * 8]);
      v16bf b0 = frag_b(&Bs[ct * 32 + low][kk + hi * 8]);
      v16bf b1 = frag_b(&Bs[ct * 32 + 16 + low][kk + hi * 8]);
      acc0 = __builtin_amdgcn_wmma_f32_16x16x32_bf16(false, a, false, b0, (short)0, acc0, false, false);
      acc1 = __builtin_amdgcn_wmma_f32_16x16x32_bf16(false, a, false, b1, (short)0, acc1, false, false);
    }
    __syncthreads();
  }
  const int c0 = gcol0 + ct * 32 + low, c1 = c0 + 16;
  const float bb0 = bias ? bias[c0] : 0.f;
  const float bb1 = bias ? bias[c1] : 0.f;
#pragma unroll
  for (int v = 0; v < 8; ++v) {
    const int r = grow0 + rt * 16 + hi * 8 + v;   // C/D layout: vgpr v -> row v (+8 for hi lanes)
    C[(size_t)r * Nc + c0] = acc0[v] + bb0;
    C[(size_t)r * Nc + c1] = acc1[v] + bb1;
  }
}

// ---------------- conversion / epilogue / irregular kernels ----------------
__global__ void k_f2bf(const float* __restrict__ s, __bf16* __restrict__ d, int n) {
  int i = blockIdx.x * blockDim.x + threadIdx.x; if (i >= n) return;
  d[i] = f2b(s[i]);
}
__global__ void k_f2bf_T(const float* __restrict__ s, __bf16* __restrict__ d, int R, int C) {
  int i = blockIdx.x * blockDim.x + threadIdx.x; if (i >= R * C) return;
  int r = i / C, c = i % C;
  d[(size_t)c * R + r] = f2b(s[i]);
}
__global__ void k_scaleRows_bf16T(const float* __restrict__ s, const float* __restrict__ sc,
                                  __bf16* __restrict__ d, int R, int C) {
  int i = blockIdx.x * blockDim.x + threadIdx.x; if (i >= R * C) return;
  int r = i / C, c = i % C;
  d[(size_t)c * R + r] = f2b(sc[r] * s[i]);
}
__global__ void k_build_A(const int* __restrict__ ei, float* __restrict__ A, float* __restrict__ deg) {
  int e = blockIdx.x * blockDim.x + threadIdx.x; if (e >= ET__) return;
  int s, d; edge_sd(ei, e, s, d);
  atomicAdd(&A[(size_t)s * N__ + d], 1.0f);
  atomicAdd(&deg[d], 1.0f);
}
__global__ void k_rsqrt(const float* __restrict__ in, float* __restrict__ out, int n) {
  int i = blockIdx.x * blockDim.x + threadIdx.x; if (i >= n) return;
  out[i] = rsqrtf(in[i]);
}
__global__ void k_gcn_ep(const float* __restrict__ agg, const float* __restrict__ dis,
                         const float* __restrict__ b, float* __restrict__ out, int R, int C) {
  int i = blockIdx.x * blockDim.x + threadIdx.x; if (i >= R * C) return;
  out[i] = dis[i / C] * agg[i] + b[i % C];
}
__global__ void k_leconv_comb(const float* __restrict__ agg, const float* __restrict__ bb,
                              const float* __restrict__ d3, const float* __restrict__ degv,
                              const float* __restrict__ b3, float* __restrict__ out, int R, int C) {
  int i = blockIdx.x * blockDim.x + threadIdx.x; if (i >= R * C) return;
  out[i] = agg[i] - degv[i / C] * bb[i] + d3[i] + b3[i % C];
}
__global__ void k_xq_max(const float* __restrict__ xp, unsigned* __restrict__ keys, const int* __restrict__ ei) {
  long long i = (long long)blockIdx.x * blockDim.x + threadIdx.x;
  if (i >= (long long)ET__ * H__) return;
  int e = (int)(i >> 9), f = (int)(i & (H__ - 1));
  int s, d; edge_sd(ei, e, s, d);
  atomicMax(&keys[(size_t)d * H__ + f], fkey(xp[(size_t)s * H__ + f]));
}
__global__ void k_decode(unsigned* __restrict__ k, int n) {
  int i = blockIdx.x * blockDim.x + threadIdx.x; if (i >= n) return;
  float v = funkey(k[i]);
  reinterpret_cast<float*>(k)[i] = v;
}
__global__ void k_matvec(const float* __restrict__ X, const float* __restrict__ v,
                         const float* __restrict__ biasp, float* __restrict__ out, int R, int C) {
  int r = blockIdx.x * blockDim.x + threadIdx.x; if (r >= R) return;
  float s = biasp ? biasp[0] : 0.f;
  for (int c = 0; c < C; ++c) s += X[(size_t)r * C + c] * v[c];
  out[r] = s;
}
__global__ void k_att_edge(const float* __restrict__ qa, const float* __restrict__ pb,
                           const float* __restrict__ ab, const int* __restrict__ ei, float* __restrict__ sc) {
  int e = blockIdx.x * blockDim.x + threadIdx.x; if (e >= ET__) return;
  int s, d; edge_sd(ei, e, s, d);
  sc[e] = lrelu(qa[d] + pb[s] + ab[0]);
}
__global__ void k_segmax_sc(const float* __restrict__ sc, unsigned* __restrict__ mk, const int* __restrict__ ei) {
  int e = blockIdx.x * blockDim.x + threadIdx.x; if (e >= ET__) return;
  int s, d; edge_sd(ei, e, s, d); (void)s;
  atomicMax(&mk[d], fkey(sc[e]));
}
__global__ void k_exp_den(float* __restrict__ sc, const float* __restrict__ m,
                          float* __restrict__ den, const int* __restrict__ ei) {
  int e = blockIdx.x * blockDim.x + threadIdx.x; if (e >= ET__) return;
  int s, d; edge_sd(ei, e, s, d); (void)s;
  float v = expf(sc[e] - m[d]);
  sc[e] = v;
  atomicAdd(&den[d], v);
}
__global__ void k_smW(const float* __restrict__ sc, const float* __restrict__ den,
                      float* __restrict__ Wm, const int* __restrict__ ei) {
  int e = blockIdx.x * blockDim.x + threadIdx.x; if (e >= ET__) return;
  int s, d; edge_sd(ei, e, s, d);
  atomicAdd(&Wm[(size_t)s * N__ + d], sc[e] / den[d]);
}
__global__ void k_edge_scalar(const float* __restrict__ a1, float* __restrict__ agg, const int* __restrict__ ei) {
  int e = blockIdx.x * blockDim.x + threadIdx.x; if (e >= ET__) return;
  int s, d; edge_sd(ei, e, s, d);
  atomicAdd(&agg[d], a1[s]);
}
__global__ void k_fitness(const float* __restrict__ agg, const float* __restrict__ degv,
                          const float* __restrict__ b2, const float* __restrict__ d3,
                          const float* __restrict__ b3, float* __restrict__ fit, int n) {
  int i = blockIdx.x * blockDim.x + threadIdx.x; if (i >= n) return;
  fit[i] = sigm(agg[i] - degv[i] * b2[i] + d3[i] + b3[0]);
}
__global__ void k_topk_rank(const float* __restrict__ v, int n, int k,
                            int* __restrict__ perm, float* __restrict__ fv) {
  int i = blockIdx.x * blockDim.x + threadIdx.x; if (i >= n) return;
  float vi = v[i];
  int rank = 0;
  for (int j = 0; j < n; ++j) {
    float vj = v[j];
    rank += (vj > vi) || (vj == vi && j < i);
  }
  if (rank < k) { perm[rank] = i; fv[rank] = vi; }
}
__global__ void k_build_Sbt(const float* __restrict__ Wm, const int* __restrict__ perm, __bf16* __restrict__ Sbt) {
  int i = blockIdx.x * blockDim.x + threadIdx.x; if (i >= K1__ * N__) return;
  int m = i >> 11, k = i & (N__ - 1);
  Sbt[i] = f2b(Wm[(size_t)k * N__ + perm[m]]);   // Sbt[m,k] = S[k,m]
}
__global__ void k_gather_scale(const float* __restrict__ X, const int* __restrict__ perm,
                               const float* __restrict__ fv, float* __restrict__ out, int Rows) {
  int i = blockIdx.x * blockDim.x + threadIdx.x; if (i >= Rows * H__) return;
  int m = i >> 9, f = i & (H__ - 1);
  out[i] = X[(size_t)perm[m] * H__ + f] * fv[m];
}
__global__ void k_colreduce(const float* __restrict__ X, int Rows, float* __restrict__ out) {
  int f = blockIdx.x * blockDim.x + threadIdx.x; if (f >= H__) return;
  float s = 0.f, mx = -1e30f;
  for (int r = 0; r < Rows; ++r) { float v = X[(size_t)r * H__ + f]; s += v; mx = fmaxf(mx, v); }
  out[f] = s / (float)Rows;
  out[H__ + f] = mx;
}
__global__ void k_zero_diag(float* __restrict__ A2) {
  int i = blockIdx.x * blockDim.x + threadIdx.x; if (i >= K1__) return;
  A2[(size_t)i * K1__ + i] = 0.f;
}
__global__ void k_build_mask(const float* __restrict__ A2, __bf16* __restrict__ MdT) {
  int i = blockIdx.x * blockDim.x + threadIdx.x; if (i >= K1__ * K1__) return;
  int c = i >> 10, r = i & (K1__ - 1);
  float v = (A2[(size_t)r * K1__ + c] != 0.f || r == c) ? 1.f : 0.f;
  MdT[i] = f2b(v);                                 // MdT[c,r] = Md[r,c]  (col-major mask)
}
__global__ void k_mask_colsum(const __bf16* __restrict__ MdT, float* __restrict__ indeg, float* __restrict__ dis2) {
  int c = blockIdx.x * blockDim.x + threadIdx.x; if (c >= K1__) return;
  float s = 0.f;
  for (int r = 0; r < K1__; ++r) s += b2f(MdT[(size_t)c * K1__ + r]);
  indeg[c] = s;
  dis2[c] = rsqrtf(s);
}
__global__ void k_masked_max(const __bf16* __restrict__ MdT, const float* __restrict__ xp2,
                             float* __restrict__ out) {
  int i = blockIdx.x * blockDim.x + threadIdx.x; if (i >= K1__ * H__) return;
  int c = i >> 9, f = i & (H__ - 1);
  float best = -1e30f;
  for (int r = 0; r < K1__; ++r)
    if (b2f(MdT[(size_t)c * K1__ + r]) != 0.f) best = fmaxf(best, xp2[(size_t)r * H__ + f]);
  out[(size_t)c * H__ + f] = best;
}
__global__ void k_sm2_stats(const __bf16* __restrict__ MdT, const float* __restrict__ pb2,
                            const float* __restrict__ qa2, const float* __restrict__ ab,
                            float* __restrict__ mx, float* __restrict__ den) {
  int c = blockIdx.x * blockDim.x + threadIdx.x; if (c >= K1__) return;
  float q = qa2[c], b = ab[0];
  float m = -1e30f;
  for (int r = 0; r < K1__; ++r)
    if (b2f(MdT[(size_t)c * K1__ + r]) != 0.f) m = fmaxf(m, lrelu(pb2[r] + q + b));
  float s = 0.f;
  for (int r = 0; r < K1__; ++r)
    if (b2f(MdT[(size_t)c * K1__ + r]) != 0.f) s += expf(lrelu(pb2[r] + q + b) - m);
  mx[c] = m; den[c] = s;
}
__global__ void k_sm2_build(const __bf16* __restrict__ MdT, const float* __restrict__ pb2,
                            const float* __restrict__ qa2, const float* __restrict__ ab,
                            const float* __restrict__ mx, const float* __restrict__ den,
                            __bf16* __restrict__ sm2T) {
  int i = blockIdx.x * blockDim.x + threadIdx.x; if (i >= K1__ * K1__) return;
  int c = i >> 10, r = i & (K1__ - 1);
  float v = 0.f;
  if (b2f(MdT[i]) != 0.f) v = expf(lrelu(pb2[r] + qa2[c] + ab[0]) - mx[c]) / den[c];
  sm2T[i] = f2b(v);                                // sm2T[c,r] = sm2[r,c]
}
__global__ void k_maskvec(const __bf16* __restrict__ MdT, const float* __restrict__ v, float* __restrict__ out) {
  int i = blockIdx.x * blockDim.x + threadIdx.x; if (i >= K1__) return;
  float s = 0.f;
  for (int k = 0; k < K1__; ++k) s += b2f(MdT[(size_t)i * K1__ + k]) * v[k];
  out[i] = s;
}
__global__ void k_head1(const float* __restrict__ r1, const float* __restrict__ r2,
                        const float* __restrict__ w, const float* __restrict__ b, float* __restrict__ hh) {
  int f = blockIdx.x * blockDim.x + threadIdx.x; if (f >= H__) return;
  float s = b[f];
  for (int j = 0; j < 2 * H__; ++j) s += (r1[j] + r2[j]) * w[(size_t)j * H__ + f];
  hh[f] = fmaxf(s, 0.f);
}
__global__ void k_head2(const float* __restrict__ hh, const float* __restrict__ w2, const float* __restrict__ b2,
                        const float* __restrict__ w3, const float* __restrict__ b3,
                        float* __restrict__ mu, float* __restrict__ ls) {
  int o = blockIdx.x * blockDim.x + threadIdx.x; if (o >= O__) return;
  float a = b2[o], b = b3[o];
  for (int f = 0; f < H__; ++f) {
    float h = hh[f];
    a += h * w2[(size_t)f * O__ + o];
    b += h * w3[(size_t)f * O__ + o];
  }
  mu[o] = a; ls[o] = b;
}
__global__ void k_tile(const float* __restrict__ mu, const float* __restrict__ ls, float* __restrict__ out) {
  int i = blockIdx.x * blockDim.x + threadIdx.x; if (i >= N__ * O__) return;
  int o = i & (O__ - 1);
  out[i] = mu[o];
  out[N__ * O__ + i] = ls[o];
}

// ---------------- host-side launch orchestration ----------------
static inline void gemm(const __bf16* A, const __bf16* Bt, float* C, const float* bias,
                        int M, int Nc, int K, hipStream_t st) {
  dim3 g(Nc / TN, M / TM);
  k_wmma_gemm<<<g, 256, 0, st>>>(A, Bt, C, bias, M, Nc, K);
}
#define L1(kern, total, ...) kern<<<((int)(((long long)(total) + 255) / 256)), 256, 0, stream>>>(__VA_ARGS__)

extern "C" void kernel_launch(void* const* d_in, const int* in_sizes, int n_in,
                              void* d_out, int out_size, void* d_ws, size_t ws_size,
                              hipStream_t stream) {
  (void)in_sizes; (void)n_in; (void)out_size; (void)ws_size;
  // inputs in setup_inputs() dict order: x, edge_index, then params in insertion order
  const float* x       = (const float*)d_in[0];
  const int*   ei      = (const int*)d_in[1];
  const float* conv1_w = (const float*)d_in[2];
  const float* conv1_b = (const float*)d_in[3];
  const float* conv2_w = (const float*)d_in[4];
  const float* conv2_b = (const float*)d_in[5];
  const float* lin1_w  = (const float*)d_in[6];
  const float* lin1_b  = (const float*)d_in[7];
  const float* lin2_w  = (const float*)d_in[8];
  const float* lin2_b  = (const float*)d_in[9];
  const float* lin3_w  = (const float*)d_in[10];
  const float* lin3_b  = (const float*)d_in[11];
  const float* P[2][14];
  for (int p = 0; p < 2; ++p)
    for (int j = 0; j < 14; ++j) P[p][j] = (const float*)d_in[12 + p * 14 + j];
  // per-pool param index: 0 lin_w, 1 lin_b, 2 att_w, 3 att_b, 4 sc_w1, 5 sc_b1,
  //                       6 sc_w2, 7 sc_w3, 8 sc_b3, 9 in_w1, 10 in_b1, 11 in_w2, 12 in_w3, 13 in_b3
  float* out = (float*)d_out;

  // ---- workspace bump allocator ----
  char* wptr = (char*)d_ws;
  auto alloc = [&](size_t bytes) -> void* {
    void* p = (void*)wptr; wptr += (bytes + 255) & ~(size_t)255; return p;
  };
  float*  A    = (float*)alloc((size_t)N__ * N__ * 4);
  float*  Wm   = (float*)alloc((size_t)N__ * N__ * 4);
  __bf16* Atb  = (__bf16*)alloc((size_t)N__ * N__ * 2);   // A col-major bf16 (= A^T row-major)
  __bf16* WmT  = (__bf16*)alloc((size_t)N__ * N__ * 2);   // Wmat col-major bf16
  __bf16* Sbt  = (__bf16*)alloc((size_t)K1__ * N__ * 2);  // S col-major bf16 (= S^T row-major)
  float*  T1   = (float*)alloc((size_t)K1__ * N__ * 4);
  __bf16* T1b  = (__bf16*)alloc((size_t)K1__ * N__ * 2);
  float*  A2   = (float*)alloc((size_t)K1__ * K1__ * 4);
  __bf16* MdT  = (__bf16*)alloc((size_t)K1__ * K1__ * 2); // mask col-major bf16 (Msl == Md)
  __bf16* sm2T = (__bf16*)alloc((size_t)K1__ * K1__ * 2);
  float* F[8];
  for (int i = 0; i < 8; ++i) F[i] = (float*)alloc((size_t)N__ * H__ * 4);
  __bf16* B0 = (__bf16*)alloc((size_t)N__ * H__ * 2);     // transposed activations [H,N] / [H,K1]
  __bf16* B1 = (__bf16*)alloc((size_t)N__ * H__ * 2);
  __bf16* B2 = (__bf16*)alloc((size_t)N__ * H__ * 2);
  __bf16* wt = (__bf16*)alloc((size_t)H__ * H__ * 2);     // current weight, transposed bf16
  __bf16* xb = (__bf16*)alloc((size_t)N__ * IN__ * 2);
  float* deg  = (float*)alloc(N__ * 4);
  float* dis  = (float*)alloc(N__ * 4);
  float* qa   = (float*)alloc(N__ * 4);
  float* pb   = (float*)alloc(N__ * 4);
  float* mbuf = (float*)alloc(N__ * 4);
  float* den  = (float*)alloc(N__ * 4);
  float* sc   = (float*)alloc(ET__ * 4);
  float* a1   = (float*)alloc(N__ * 4);
  float* b2v  = (float*)alloc(N__ * 4);
  float* d3v  = (float*)alloc(N__ * 4);
  float* agg1 = (float*)alloc(N__ * 4);
  float* fit  = (float*)alloc(N__ * 4);
  int*   perm = (int*)alloc(K1__ * 4);
  float* fv   = (float*)alloc(K1__ * 4);
  float* indeg= (float*)alloc(K1__ * 4);
  float* dis2 = (float*)alloc(K1__ * 4);
  float* pb2  = (float*)alloc(K1__ * 4);
  float* qa2  = (float*)alloc(K1__ * 4);
  float* mx2  = (float*)alloc(K1__ * 4);
  float* den2 = (float*)alloc(K1__ * 4);
  float* t1v  = (float*)alloc(K1__ * 4);
  float* fit2 = (float*)alloc(K1__ * 4);
  int*   perm2= (int*)alloc(K2__ * 4);
  float* fv2  = (float*)alloc(K2__ * 4);
  float* r1   = (float*)alloc(2 * H__ * 4);
  float* r2   = (float*)alloc(2 * H__ * 4);
  float* hh   = (float*)alloc(H__ * 4);
  float* mu   = (float*)alloc(O__ * 4);
  float* ls   = (float*)alloc(O__ * 4);

  // ================= conv1 (GCN): h1 = dis * (A^T (dis*(x W))) + b =================
  hipMemsetAsync(A, 0, (size_t)N__ * N__ * 4, stream);
  hipMemsetAsync(deg, 0, N__ * 4, stream);
  L1(k_build_A, ET__, ei, A, deg);
  L1(k_rsqrt, N__, deg, dis, N__);
  L1(k_f2bf, N__ * IN__, x, xb, N__ * IN__);
  L1(k_f2bf_T, IN__ * H__, conv1_w, wt, IN__, H__);
  gemm(xb, wt, F[0], nullptr, N__, H__, IN__, stream);               // h = x @ W
  L1(k_f2bf_T, (long long)N__ * N__, A, Atb, N__, N__);              // A^T bf16
  L1(k_scaleRows_bf16T, N__ * H__, F[0], dis, B0, N__, H__);         // (dis*h)^T bf16
  gemm(Atb, B0, F[7], nullptr, N__, H__, N__, stream);               // A^T (dis*h)
  L1(k_gcn_ep, N__ * H__, F[7], dis, conv1_b, F[1], N__, H__);       // h1

  // ================= pool1: LEConv -> xp =================
  L1(k_f2bf, N__ * H__, F[1], B1, N__ * H__);                        // h1 bf16
  L1(k_f2bf_T, H__ * H__, P[0][9], wt, H__, H__);
  gemm(B1, wt, F[5], P[0][10], N__, H__, H__, stream);               // a = h1 w1 + b1
  L1(k_f2bf_T, H__ * H__, P[0][11], wt, H__, H__);
  gemm(B1, wt, F[6], nullptr, N__, H__, H__, stream);                // bb = h1 w2
  L1(k_f2bf_T, H__ * H__, P[0][12], wt, H__, H__);
  gemm(B1, wt, F[0], nullptr, N__, H__, H__, stream);                // d3 = h1 w3
  L1(k_f2bf_T, N__ * H__, F[5], B0, N__, H__);                       // a^T
  gemm(Atb, B0, F[7], nullptr, N__, H__, N__, stream);               // A^T a
  L1(k_leconv_comb, N__ * H__, F[7], F[6], F[0], deg, P[0][13], F[2], N__, H__); // xp

  // xq = segment_max(xp[src], dst) @ lin + b
  hipMemsetAsync(F[3], 0, (size_t)N__ * H__ * 4, stream);
  L1(k_xq_max, (long long)ET__ * H__, F[2], (unsigned*)F[3], ei);
  L1(k_decode, N__ * H__, (unsigned*)F[3], N__ * H__);
  L1(k_f2bf, N__ * H__, F[3], B2, N__ * H__);
  L1(k_f2bf_T, H__ * H__, P[0][0], wt, H__, H__);
  gemm(B2, wt, F[4], P[0][1], N__, H__, H__, stream);                // xq

  // attention softmax over edges -> Wmat (weighted adjacency)
  L1(k_matvec, N__, F[4], P[0][2], nullptr, qa, N__, H__);           // xq @ wa
  L1(k_matvec, N__, F[2], P[0][2] + H__, nullptr, pb, N__, H__);     // xp @ wb
  L1(k_att_edge, ET__, qa, pb, P[0][3], ei, sc);
  hipMemsetAsync(mbuf, 0, N__ * 4, stream);
  L1(k_segmax_sc, ET__, sc, (unsigned*)mbuf, ei);
  L1(k_decode, N__, (unsigned*)mbuf, N__);
  hipMemsetAsync(den, 0, N__ * 4, stream);
  L1(k_exp_den, ET__, sc, mbuf, den, ei);
  hipMemsetAsync(Wm, 0, (size_t)N__ * N__ * 4, stream);
  L1(k_smW, ET__, sc, den, Wm, ei);

  // xc = Wmat^T @ h1
  L1(k_f2bf_T, (long long)N__ * N__, Wm, WmT, N__, N__);
  L1(k_f2bf_T, N__ * H__, F[1], B0, N__, H__);                       // h1^T
  gemm(WmT, B0, F[5], nullptr, N__, H__, N__, stream);               // xc

  // fitness + top_k
  L1(k_matvec, N__, F[5], P[0][4], P[0][5], a1, N__, H__);
  L1(k_matvec, N__, F[5], P[0][6], nullptr, b2v, N__, H__);
  L1(k_matvec, N__, F[5], P[0][7], nullptr, d3v, N__, H__);
  hipMemsetAsync(agg1, 0, N__ * 4, stream);
  L1(k_edge_scalar, ET__, a1, agg1, ei);
  L1(k_fitness, N__, agg1, deg, b2v, d3v, P[0][8], fit, N__);
  L1(k_topk_rank, N__, fit, N__, K1__, perm, fv);
  L1(k_gather_scale, K1__ * H__, F[5], perm, fv, F[6], K1__);        // xn
  L1(k_colreduce, H__, F[6], K1__, r1);

  // coarsening: A2 = (S^T A) S with zero diagonal
  L1(k_build_Sbt, K1__ * N__, Wm, perm, Sbt);
  gemm(Sbt, Atb, T1, nullptr, K1__, N__, N__, stream);               // T1 = S^T A
  L1(k_f2bf, K1__ * N__, T1, T1b, K1__ * N__);
  gemm(T1b, Sbt, A2, nullptr, K1__, K1__, N__, stream);              // A2 = T1 S
  L1(k_zero_diag, K1__, A2);

  // mask (Msl == Md), indeg & dis2
  L1(k_build_mask, K1__ * K1__, A2, MdT);
  L1(k_mask_colsum, K1__, MdT, indeg, dis2);

  // ================= conv2: h2 = dis2*(Md^T (dis2*(xn W))) + b =================
  L1(k_f2bf, K1__ * H__, F[6], B1, K1__ * H__);                      // xn bf16
  L1(k_f2bf_T, H__ * H__, conv2_w, wt, H__, H__);
  gemm(B1, wt, F[0], nullptr, K1__, H__, H__, stream);               // g = xn W
  L1(k_scaleRows_bf16T, K1__ * H__, F[0], dis2, B0, K1__, H__);      // (dis2*g)^T
  gemm(MdT, B0, F[7], nullptr, K1__, H__, K1__, stream);             // Md^T (dis2*g)
  L1(k_gcn_ep, K1__ * H__, F[7], dis2, conv2_b, F[1], K1__, H__);    // h2

  // ================= pool2 (dense) =================
  L1(k_f2bf, K1__ * H__, F[1], B1, K1__ * H__);                      // h2 bf16
  L1(k_f2bf_T, H__ * H__, P[1][9], wt, H__, H__);
  gemm(B1, wt, F[5], P[1][10], K1__, H__, H__, stream);              // a2 = h2 w1 + b1
  L1(k_f2bf_T, H__ * H__, P[1][11], wt, H__, H__);
  gemm(B1, wt, F[6], nullptr, K1__, H__, H__, stream);               // c2 = h2 w2
  L1(k_f2bf_T, H__ * H__, P[1][12], wt, H__, H__);
  gemm(B1, wt, F[0], nullptr, K1__, H__, H__, stream);               // d2 = h2 w3
  L1(k_f2bf_T, K1__ * H__, F[5], B0, K1__, H__);                     // a2^T
  gemm(MdT, B0, F[7], nullptr, K1__, H__, K1__, stream);             // Md^T a2
  L1(k_leconv_comb, K1__ * H__, F[7], F[6], F[0], indeg, P[1][13], F[2], K1__, H__); // xp2

  L1(k_masked_max, K1__ * H__, MdT, F[2], F[3]);                     // masked max
  L1(k_f2bf, K1__ * H__, F[3], B2, K1__ * H__);
  L1(k_f2bf_T, H__ * H__, P[1][0], wt, H__, H__);
  gemm(B2, wt, F[4], P[1][1], K1__, H__, H__, stream);               // xq2

  L1(k_matvec, K1__, F[2], P[1][2] + H__, nullptr, pb2, K1__, H__);  // xp2 @ wb2
  L1(k_matvec, K1__, F[4], P[1][2], nullptr, qa2, K1__, H__);        // xq2 @ wa2
  L1(k_sm2_stats, K1__, MdT, pb2, qa2, P[1][3], mx2, den2);
  L1(k_sm2_build, K1__ * K1__, MdT, pb2, qa2, P[1][3], mx2, den2, sm2T);
  L1(k_f2bf_T, K1__ * H__, F[1], B0, K1__, H__);                     // h2^T
  gemm(sm2T, B0, F[5], nullptr, K1__, H__, K1__, stream);            // xc2 = sm2^T h2

  L1(k_matvec, K1__, F[5], P[1][4], P[1][5], a1, K1__, H__);
  L1(k_matvec, K1__, F[5], P[1][6], nullptr, b2v, K1__, H__);
  L1(k_matvec, K1__, F[5], P[1][7], nullptr, d3v, K1__, H__);
  L1(k_maskvec, K1__, MdT, a1, t1v);
  L1(k_fitness, K1__, t1v, indeg, b2v, d3v, P[1][8], fit2, K1__);
  L1(k_topk_rank, K1__, fit2, K1__, K2__, perm2, fv2);
  L1(k_gather_scale, K2__ * H__, F[5], perm2, fv2, F[6], K2__);      // xn2
  L1(k_colreduce, H__, F[6], K2__, r2);

  // ================= head =================
  L1(k_head1, H__, r1, r2, lin1_w, lin1_b, hh);
  L1(k_head2, O__, hh, lin2_w, lin2_b, lin3_w, lin3_b, mu, ls);
  L1(k_tile, N__ * O__, mu, ls, out);
}
